// CTCLoss_15633680958143
// MI455X (gfx1250) — matrix operand order repaired and checked by reading, not compiled, not run on hardware
//
#include <hip/hip_runtime.h>
#include <math.h>

// Problem constants (fixed by the reference's setup_inputs)
#define Bsz  16
#define Tlen 512
#define Hdim 512
#define Vdim 4096
#define Llen 64
#define Sext 129   // 2L+1 extended label length
#define SP   132   // padded S for alignment

#define NEGINF (-1e30f)

typedef __attribute__((ext_vector_type(8)))  float  v8f;
typedef __attribute__((ext_vector_type(8)))  __bf16 v8bf;
typedef __attribute__((ext_vector_type(16))) __bf16 v16bf;

// gfx1250 async global->LDS path (ASYNCcnt), guarded so missing builtins fall
// back to the synchronous staging path. The builtin's probed signature is
// (int4_asd(1)*, int4_as(3)*, imm int offset, imm int cpol) for the b128 form.
#if defined(__gfx1250__) && __has_builtin(__builtin_amdgcn_global_load_async_to_lds_b128)
#define HAVE_ASYNC_LDS 1
typedef int v4i_g __attribute__((__vector_size__(4 * sizeof(int))));
typedef __attribute__((address_space(1))) v4i_g as1_v4i;
typedef __attribute__((address_space(3))) v4i_g as3_v4i;
#else
#define HAVE_ASYNC_LDS 0
#endif

static __device__ __forceinline__ void wait_asynccnt0() {
#if __has_builtin(__builtin_amdgcn_s_wait_asynccnt)
  __builtin_amdgcn_s_wait_asynccnt(0);
#else
  asm volatile("s_wait_asynccnt 0" ::: "memory");
#endif
}

static __device__ __forceinline__ v16bf join16(v8bf a, v8bf b) {
  return __builtin_shufflevector(a, b, 0,1,2,3,4,5,6,7,8,9,10,11,12,13,14,15);
}

static __device__ __forceinline__ float lae(float x, float y) {
  float m = fmaxf(x, y);
  float d = fminf(x, y) - m;      // <= 0
  return m + log1pf(__expf(d));
}

// ---------------------------------------------------------------------------
// Kernel 1: transpose W [H,V] f32 -> Wt[V][H] bf16 split into hi + lo parts.
// LDS-tiled so both the global read and write are coalesced.
// ---------------------------------------------------------------------------
__global__ __launch_bounds__(256) void wsplit_kernel(const float* __restrict__ W,
                                                     unsigned short* __restrict__ WtHi,
                                                     unsigned short* __restrict__ WtLo) {
  __shared__ float tile[32][33];
  int v0 = blockIdx.x * 32;
  int h0 = blockIdx.y * 32;
  int tx = threadIdx.x, ty = threadIdx.y;
  #pragma unroll
  for (int i = 0; i < 4; ++i) {
    int h = h0 + ty + 8 * i;
    tile[ty + 8 * i][tx] = W[(size_t)h * Vdim + (v0 + tx)];
  }
  __syncthreads();
  #pragma unroll
  for (int i = 0; i < 4; ++i) {
    int v = v0 + ty + 8 * i;
    int h = h0 + tx;
    float x = tile[tx][ty + 8 * i];
    __bf16 hi = (__bf16)x;
    __bf16 lo = (__bf16)(x - (float)hi);
    size_t idx = (size_t)v * Hdim + h;
    WtHi[idx] = __builtin_bit_cast(unsigned short, hi);
    WtLo[idx] = __builtin_bit_cast(unsigned short, lo);
  }
}

// ---------------------------------------------------------------------------
// Kernel 2: extended-label table + pre-gather of label columns of W (so the
// emit kernel reads contiguous vectors) and gathered bias.
// ---------------------------------------------------------------------------
__global__ __launch_bounds__(128) void gather_kernel(const int* __restrict__ ys,
                                                     const float* __restrict__ W,
                                                     const float* __restrict__ bias,
                                                     int* __restrict__ ext,
                                                     float* __restrict__ Wg,
                                                     float* __restrict__ biasg) {
  int bs = blockIdx.x;
  int b = bs / Sext, s = bs % Sext;
  int v = (s & 1) ? ys[b * Llen + ((s - 1) >> 1)] : 0;   // blank=0 at even s
  if (threadIdx.x == 0) {
    ext[b * SP + s] = v;
    biasg[b * SP + s] = bias[v];
  }
  float* dst = Wg + ((size_t)b * SP + s) * Hdim;
  for (int h = threadIdx.x; h < Hdim; h += 128)
    dst[h] = W[(size_t)h * Vdim + v];
}

// ---------------------------------------------------------------------------
// Kernel 3: WMMA GEMM (bf16x2 split, f32 accumulate) fused with an online
// logsumexp over V. One WG = (batch b, 16 time rows); 8 waves each own 512
// columns. lse[b][t] = log sum_v exp(hs[b,t]·W[:,v] + bias[v]).
// A tile is staged via the gfx1250 async global->LDS engine when available.
// ---------------------------------------------------------------------------
__global__ __launch_bounds__(256) void gemm_lse_kernel(const float* __restrict__ hs,
                                                       const unsigned short* __restrict__ WtHi,
                                                       const unsigned short* __restrict__ WtLo,
                                                       const float* __restrict__ bias,
                                                       float* __restrict__ lse) {
  // A tile (16 x 512) in LDS, bf16 hi/lo; stride 520 (=65*16B rows) keeps
  // ds_load_b128 fragments 16B-aligned and bank-conflict free across lanes.
  __shared__ __align__(16) __bf16 Ahi[16][520];
  __shared__ __align__(16) __bf16 Alo[16][520];
  __shared__ float redM[8][16];
  __shared__ float redS[8][16];
#if HAVE_ASYNC_LDS
  __shared__ __align__(16) float stageA[16][Hdim];
#endif

  int b  = blockIdx.x >> 5;          // 16 batches
  int t0 = (blockIdx.x & 31) << 4;   // 32 time-tiles of 16

#if HAVE_ASYNC_LDS
  // Async-copy the raw f32 hs tile (32 KB) straight into LDS (ASYNCcnt path),
  // then convert to bf16 hi/lo in LDS.
  for (int i = threadIdx.x; i < 16 * (Hdim / 4); i += 256) {
    int m  = i >> 7;                 // 128 float4 per row
    int k4 = i & 127;
    const float* g = &hs[((size_t)b * Tlen + t0 + m) * Hdim + k4 * 4];
    __builtin_amdgcn_global_load_async_to_lds_b128(
        (as1_v4i*)g,
        (as3_v4i*)&stageA[m][k4 * 4],
        0, 0);
  }
  wait_asynccnt0();                  // own async loads complete
  __syncthreads();                   // everyone's loads complete
  for (int i = threadIdx.x; i < 16 * Hdim; i += 256) {
    int m = i >> 9, k = i & (Hdim - 1);
    float x = stageA[m][k];
    __bf16 hi = (__bf16)x;
    Ahi[m][k] = hi;
    Alo[m][k] = (__bf16)(x - (float)hi);
  }
#else
  for (int i = threadIdx.x; i < 16 * Hdim; i += 256) {
    int m = i >> 9, k = i & (Hdim - 1);
    float x = hs[((size_t)b * Tlen + t0 + m) * Hdim + k];
    __bf16 hi = (__bf16)x;
    Ahi[m][k] = hi;
    Alo[m][k] = (__bf16)(x - (float)hi);
  }
#endif
  __syncthreads();

  int wave = threadIdx.x >> 5;
  int lane = threadIdx.x & 31;
  int nl   = lane & 15;              // N (B/C/D) or M (A) index within half
  int half = lane >> 4;

  float mRun[8], sRun[8];            // online softmax state, one per C VGPR (row)
  #pragma unroll
  for (int r = 0; r < 8; ++r) { mRun[r] = NEGINF; sRun[r] = 0.f; }

  const __bf16* arowHi = &Ahi[nl][0];
  const __bf16* arowLo = &Alo[nl][0];

  for (int nt = 0; nt < 32; ++nt) {
    int col = wave * 512 + nt * 16 + nl;
    float bv = bias[col];
    v8f acc;
    #pragma unroll
    for (int r = 0; r < 8; ++r) acc[r] = bv;   // fold bias into accumulator

    const __bf16* bhiP = (const __bf16*)WtHi + (size_t)col * Hdim + half * 16;
    const __bf16* bloP = (const __bf16*)WtLo + (size_t)col * Hdim + half * 16;

    #pragma unroll 4
    for (int k0 = 0; k0 < Hdim; k0 += 32) {
      __builtin_prefetch(bhiP + k0 + 64, 0, 1);   // global_prefetch_b8
      __builtin_prefetch(bloP + k0 + 64, 0, 1);
      // A fragment (lanes<16 K=0..15, lanes>=16 K=16..31; M=lane&15)
      v8bf ah0 = *(const v8bf*)(arowHi + k0 + half * 16);
      v8bf ah1 = *(const v8bf*)(arowHi + k0 + half * 16 + 8);
      v8bf al0 = *(const v8bf*)(arowLo + k0 + half * 16);
      v8bf al1 = *(const v8bf*)(arowLo + k0 + half * 16 + 8);
      // B fragment (lanes<16 K=0..15, lanes>=16 K=16..31; N=lane&15)
      v8bf bh0 = *(const v8bf*)(bhiP + k0);
      v8bf bh1 = *(const v8bf*)(bhiP + k0 + 8);
      v8bf bl0 = *(const v8bf*)(bloP + k0);
      v8bf bl1 = *(const v8bf*)(bloP + k0 + 8);
      v16bf aHi = join16(ah0, ah1), aLo = join16(al0, al1);
      v16bf bHi = join16(bh0, bh1), bLo = join16(bl0, bl1);
      // bf16x2 split product: hi*hi + hi*lo + lo*hi (~fp32 quality)
      acc = __builtin_amdgcn_wmma_f32_16x16x32_bf16(false, aHi, false, bHi, (short)0, acc, false, false);
      acc = __builtin_amdgcn_wmma_f32_16x16x32_bf16(false, aHi, false, bLo, (short)0, acc, false, false);
      acc = __builtin_amdgcn_wmma_f32_16x16x32_bf16(false, aLo, false, bHi, (short)0, acc, false, false);
    }

    // Online logsumexp update: each lane holds one column of 8 rows.
    #pragma unroll
    for (int r = 0; r < 8; ++r) {
      float v  = acc[r];
      float mo = mRun[r];
      float mn = fmaxf(mo, v);
      sRun[r] = sRun[r] * __expf(mo - mn) + __expf(v - mn);
      mRun[r] = mn;
    }
  }

  // Reduce across the 16 lanes of each half (rows r / 8+r live in separate halves).
  #pragma unroll
  for (int off = 1; off < 16; off <<= 1) {
    #pragma unroll
    for (int r = 0; r < 8; ++r) {
      float mo = __shfl_xor(mRun[r], off, 32);
      float so = __shfl_xor(sRun[r], off, 32);
      float mn = fmaxf(mRun[r], mo);
      sRun[r] = sRun[r] * __expf(mRun[r] - mn) + so * __expf(mo - mn);
      mRun[r] = mn;
    }
  }
  if (nl == 0) {
    #pragma unroll
    for (int r = 0; r < 8; ++r) {
      redM[wave][half * 8 + r] = mRun[r];
      redS[wave][half * 8 + r] = sRun[r];
    }
  }
  __syncthreads();
  if (threadIdx.x < 16) {
    int row = threadIdx.x;
    float m = NEGINF, s = 0.f;
    #pragma unroll
    for (int w = 0; w < 8; ++w) {
      float mo = redM[w][row], so = redS[w][row];
      float mn = fmaxf(m, mo);
      s = s * __expf(m - mn) + so * __expf(mo - mn);
      m = mn;
    }
    lse[b * Tlen + t0 + row] = m + __logf(s);
  }
}

// ---------------------------------------------------------------------------
// Kernel 4: emission log-probs at extended-label positions only:
// emit[t][b][s] = hs[b,t]·Wg[b,s] + biasg[b,s] - lse[b,t]   (fp32, ~1 GFLOP)
// ---------------------------------------------------------------------------
__global__ __launch_bounds__(160) void emit_kernel(const float* __restrict__ hs,
                                                   const float* __restrict__ Wg,
                                                   const float* __restrict__ biasg,
                                                   const float* __restrict__ lse,
                                                   float* __restrict__ emit) {
  __shared__ __align__(16) float hrow[Hdim];
  int b = blockIdx.x / Tlen;
  int t = blockIdx.x % Tlen;
  const float* src = hs + ((size_t)b * Tlen + t) * Hdim;
  for (int i = threadIdx.x; i < Hdim; i += 160) hrow[i] = src[i];
  __syncthreads();
  int s = threadIdx.x;
  if (s < Sext) {
    const float4* wc = (const float4*)(Wg + ((size_t)b * SP + s) * Hdim);
    const float4* hr = (const float4*)hrow;
    float acc = 0.f;
    #pragma unroll 8
    for (int i = 0; i < Hdim / 4; ++i) {
      float4 a = wc[i], c = hr[i];
      acc += a.x * c.x + a.y * c.y + a.z * c.z + a.w * c.w;
    }
    emit[((size_t)t * Bsz + b) * SP + s] = acc + biasg[b * SP + s] - lse[b * Tlen + t];
  }
}

// ---------------------------------------------------------------------------
// Kernel 5: CTC alpha DP. One WG, 16 waves = 16 batches. Alpha in LDS with a
// 2-slot NEG pad in front; each wave covers S=129 states in 5 lane-chunks.
// Read-phase then write-phase per t (same-wave LDS ops are in program order,
// so no barriers needed inside the T loop). Deterministic final sum via LDS.
// ---------------------------------------------------------------------------
__global__ __launch_bounds__(512) void ctc_kernel(const float* __restrict__ emit,
                                                  const int* __restrict__ ext,
                                                  const int* __restrict__ in_lens,
                                                  const int* __restrict__ lab_lens,
                                                  float* __restrict__ out) {
  __shared__ float alpha[Bsz][SP + 4];
  __shared__ float partial[Bsz];
  int wave = threadIdx.x >> 5;
  int lane = threadIdx.x & 31;
  int b = wave;
  float* A = &alpha[b][0];          // A[0..1] = NEG pad; A[2+s] = alpha_s

  for (int i = lane; i < SP + 4; i += 32) A[i] = NEGINF;
  if (lane == 0) {                  // alpha0
    A[2] = emit[(size_t)b * SP + 0];
    A[3] = emit[(size_t)b * SP + 1];
  }

  // Per-chunk skip flags (s>=2 && ext[s]!=blank && ext[s]!=ext[s-2])
  bool skipF[5];
  #pragma unroll
  for (int c = 0; c < 5; ++c) {
    int s = lane + 32 * c;
    bool sk = false;
    if (s >= 2 && s < Sext) {
      int e  = ext[b * SP + s];
      int e2 = ext[b * SP + s - 2];
      sk = (e != 0) && (e != e2);
    }
    skipF[c] = sk;
  }
  int inlen = in_lens[b];

  for (int t = 1; t < inlen; ++t) {
    const float* em = emit + ((size_t)t * Bsz + b) * SP;
    float nv[5] = {0.f, 0.f, 0.f, 0.f, 0.f};
    #pragma unroll
    for (int c = 0; c < 5; ++c) {       // read phase
      int s = lane + 32 * c;
      if (s < Sext) {
        float a0 = A[2 + s], a1 = A[1 + s], a2 = A[s];
        float comb = lae(a0, a1);
        if (skipF[c]) comb = lae(comb, a2);
        nv[c] = comb + em[s];
      }
    }
    #pragma unroll
    for (int c = 0; c < 5; ++c) {       // write phase
      int s = lane + 32 * c;
      if (s < Sext) A[2 + s] = nv[c];
    }
  }

  if (lane == 0) {
    int sl = 2 * lab_lens[b];
    partial[b] = -lae(A[2 + sl], A[1 + sl]);
  }
  __syncthreads();
  if (threadIdx.x == 0) {
    float s = 0.f;
    #pragma unroll
    for (int i = 0; i < Bsz; ++i) s += partial[i];
    out[0] = s / (float)Bsz;
  }
}

// ---------------------------------------------------------------------------
extern "C" void kernel_launch(void* const* d_in, const int* in_sizes, int n_in,
                              void* d_out, int out_size, void* d_ws, size_t ws_size,
                              hipStream_t stream) {
  const float* hs      = (const float*)d_in[0];   // [B,T,H] f32
  const int*   hs_lens = (const int*)d_in[1];     // [B]
  const int*   ys      = (const int*)d_in[2];     // [B,L]
  const int*   ys_lens = (const int*)d_in[3];     // [B]
  const float* W       = (const float*)d_in[4];   // [H,V] f32
  const float* bias    = (const float*)d_in[5];   // [V]
  float* out = (float*)d_out;
  (void)in_sizes; (void)n_in; (void)out_size; (void)ws_size;

  // Workspace carve (~17 MB total)
  char* ws = (char*)d_ws;
  size_t off = 0;
  auto carve = [&](size_t bytes) -> char* {
    char* p = ws + off;
    off = (off + bytes + 255) & ~(size_t)255;
    return p;
  };
  unsigned short* WtHi = (unsigned short*)carve((size_t)Vdim * Hdim * 2);
  unsigned short* WtLo = (unsigned short*)carve((size_t)Vdim * Hdim * 2);
  float* lse   = (float*)carve((size_t)Bsz * Tlen * 4);
  float* emit  = (float*)carve((size_t)Tlen * Bsz * SP * 4);
  int*   ext   = (int*)carve((size_t)Bsz * SP * 4);
  float* biasg = (float*)carve((size_t)Bsz * SP * 4);
  float* Wg    = (float*)carve((size_t)Bsz * SP * Hdim * 4);

  wsplit_kernel<<<dim3(Vdim / 32, Hdim / 32), dim3(32, 8), 0, stream>>>(W, WtHi, WtLo);
  gather_kernel<<<dim3(Bsz * Sext), dim3(128), 0, stream>>>(ys, W, bias, ext, Wg, biasg);
  gemm_lse_kernel<<<dim3(Bsz * (Tlen / 16)), dim3(256), 0, stream>>>(hs, WtHi, WtLo, bias, lse);
  emit_kernel<<<dim3(Bsz * Tlen), dim3(160), 0, stream>>>(hs, Wg, biasg, lse, emit);
  ctc_kernel<<<dim3(1), dim3(512), 0, stream>>>(emit, ext, hs_lens, ys_lens, out);
}
